// ContinuousFilterConv_65678639891011
// MI455X (gfx1250) — matrix operand order repaired and checked by reading, not compiled, not run on hardware
//
#include <hip/hip_runtime.h>
#include <hip/hip_bf16.h>

#define BDIM 4
#define ADIM 2048
#define NDIM 64
#define GDIM 64
#define FDIM 128

typedef __attribute__((ext_vector_type(16))) _Float16 v16h;
typedef __attribute__((ext_vector_type(8)))  _Float16 v8h;
typedef __attribute__((ext_vector_type(8)))  float    v8f;

// LDS partition (bytes) -- single dynamic allocation, base offset 0
#define OFF_W2T   0                      // 128*128 f16 = 32768
#define OFF_W1T   32768                  // 128*64  f16 = 16384
#define OFF_A1    49152                  // 64*64   f16 =  8192
#define OFF_HID   57344                  // 64*128  f16 = 16384
#define OFF_NF    73728                  // 64*128  f32 = 32768
#define OFF_B1    106496                 // 128 f32
#define OFF_B2    107008                 // 128 f32
#define OFF_NFW   107520                 // 128 f32
#define OFF_LOGIT 108032                 // 64 f32
#define OFF_ATT   108288                 // 64 f32
#define OFF_OUT   108544                 // 128 f32
#define OFF_NBR   109056                 // 64 i32
#define SMEM_BYTES 109312

// native CDNA5 tanh (V_TANH_F32) instead of libm expansion
__device__ __forceinline__ float fast_tanh(float x) {
#if __has_builtin(__builtin_amdgcn_tanhf)
  return __builtin_amdgcn_tanhf(x);
#else
  float r;
  asm("v_tanh_f32 %0, %1" : "=v"(r) : "v"(x));
  return r;
#endif
}

// A-matrix 16x32 f16 fragment (doc layout: lanes0-15 K 0-7|16-23, lanes16-31 K 8-15|24-31)
__device__ __forceinline__ v16h load_a_frag(const _Float16* row_ptr, int kbase, int hi) {
  const _Float16* p = row_ptr + kbase + hi * 8;
  v8h lo = *(const v8h*)(p);
  v8h hh = *(const v8h*)(p + 16);
  v16h r;
#pragma unroll
  for (int i = 0; i < 8; ++i) { r[i] = lo[i]; r[i + 8] = hh[i]; }
  return r;
}

// B-matrix 32x16 f16 fragment (lanes0-15 K 0-7|16-23, lanes16-31 K 8-15|24-31, matches A)
__device__ __forceinline__ v16h load_b_frag(const _Float16* col_ptr, int kbase, int hi) {
  const _Float16* p = col_ptr + kbase + hi * 8;
  v8h lo = *(const v8h*)(p);
  v8h hh = *(const v8h*)(p + 16);
  v16h r;
#pragma unroll
  for (int i = 0; i < 8; ++i) { r[i] = lo[i]; r[i + 8] = hh[i]; }
  return r;
}

__launch_bounds__(256)
__global__ void cfconv_kernel(const float* __restrict__ feats,
                              const float* __restrict__ rbf,
                              const long long* __restrict__ nbrs,
                              const float* __restrict__ W1, const float* __restrict__ b1,
                              const float* __restrict__ W2, const float* __restrict__ b2,
                              const float* __restrict__ nfw,
                              float* __restrict__ out, float* __restrict__ att_out) {
  extern __shared__ char smem[];
  _Float16* sW2t = (_Float16*)(smem + OFF_W2T);   // [F][F] = [f][k]
  _Float16* sW1t = (_Float16*)(smem + OFF_W1T);   // [F][G] = [f][g]
  _Float16* sA1  = (_Float16*)(smem + OFF_A1);    // [N][G]
  _Float16* sHid = (_Float16*)(smem + OFF_HID);   // [N][F]
  float*    sNF  = (float*)(smem + OFF_NF);       // [N][F]
  float*    sB1  = (float*)(smem + OFF_B1);
  float*    sB2  = (float*)(smem + OFF_B2);
  float*    sNFw = (float*)(smem + OFF_NFW);
  float*    sLog = (float*)(smem + OFF_LOGIT);
  float*    sAtt = (float*)(smem + OFF_ATT);
  float*    sOut = (float*)(smem + OFF_OUT);
  int*      sNbr = (int*)(smem + OFF_NBR);

  const int tid  = threadIdx.x;
  const int lane = tid & 31;
  const int wave = tid >> 5;
  const int ba   = blockIdx.x;                     // b*A + a
  const int b    = ba >> 11;                       // A = 2048
  const size_t featBase = (size_t)b * ADIM * FDIM;

  // ---- phase 1: stage weights, rbf tile, biases, neighbor ids ----
  for (int i = tid; i < GDIM * FDIM; i += 256) {   // W1[g][f] -> sW1t[f][g]
    int g = i >> 7, f = i & 127;
    sW1t[f * GDIM + g] = (_Float16)W1[i];
  }
  for (int i = tid; i < FDIM * FDIM; i += 256) {   // W2[k][f] -> sW2t[f][k]
    int k = i >> 7, f = i & 127;
    sW2t[f * FDIM + k] = (_Float16)W2[i];
  }
  {
    const float* rb = rbf + (size_t)ba * (NDIM * GDIM);
    for (int i = tid; i < NDIM * GDIM; i += 256) sA1[i] = (_Float16)rb[i];
  }
  if (tid < FDIM) {
    sB1[tid] = b1[tid]; sB2[tid] = b2[tid]; sNFw[tid] = nfw[tid]; sOut[tid] = 0.0f;
  }
  if (tid < NDIM) {
    sNbr[tid] = (int)nbrs[(size_t)ba * NDIM + tid];
    sLog[tid] = 0.0f;
  }
  __syncthreads();

  // ---- phase 2: async gather neighbor features global->LDS (overlaps GEMM1) ----
  {
    const float* gbase = feats + featBase;
#pragma unroll
    for (int it = 0; it < (NDIM * FDIM) / (4 * 256); ++it) {   // 2048 16B chunks / 256 thr
      int c = tid + it * 256;
      int row = c >> 5;                         // 32 chunks per 128-float row
      int cir = c & 31;
      unsigned long long ga =
          (unsigned long long)(const void*)(gbase + (size_t)sNbr[row] * FDIM + cir * 4);
      unsigned int ldsoff = OFF_NF + (unsigned int)c * 16u;
      asm volatile("global_load_async_to_lds_b128 %0, %1, off"
                   :: "v"(ldsoff), "v"(ga)
                   : "memory");
    }
  }

  // wave tile: rows m0..m0+15, cols c0..c0+63
  const int m0  = (wave & 3) * 16;
  const int c0  = (wave >> 2) * 64;
  const int hi  = (lane >> 4) & 1;
  const int l15 = lane & 15;
  const int arow = m0 + l15;

  // ---- GEMM1: hidden = tanh(rbf @ W1 + b1) ----
  v8f h[4];
#pragma unroll
  for (int nt = 0; nt < 4; ++nt) {
    float bb = sB1[c0 + nt * 16 + l15];
#pragma unroll
    for (int r = 0; r < 8; ++r) h[nt][r] = bb;
  }
#pragma unroll
  for (int kk = 0; kk < GDIM; kk += 32) {
    v16h af = load_a_frag(sA1 + arow * GDIM, kk, hi);
#pragma unroll
    for (int nt = 0; nt < 4; ++nt) {
      int n = c0 + nt * 16 + l15;
      v16h bf = load_b_frag(sW1t + n * GDIM, kk, hi);
      h[nt] = __builtin_amdgcn_wmma_f32_16x16x32_f16(false, af, false, bf,
                                                     (short)0, h[nt], false, false);
    }
  }
#pragma unroll
  for (int nt = 0; nt < 4; ++nt) {
    int col = c0 + nt * 16 + l15;
#pragma unroll
    for (int r = 0; r < 8; ++r) {
      int row = m0 + r + 8 * hi;
      sHid[row * FDIM + col] = (_Float16)fast_tanh(h[nt][r]);
    }
  }
  // async gather must be complete (and visible after the barrier) before sNF reads
  asm volatile("s_wait_asynccnt 0x0" ::: "memory");
  __syncthreads();

  // ---- GEMM2: filters = hidden @ W2 + b2 ----
  v8f fc[4];
#pragma unroll
  for (int nt = 0; nt < 4; ++nt) {
    float bb = sB2[c0 + nt * 16 + l15];
#pragma unroll
    for (int r = 0; r < 8; ++r) fc[nt][r] = bb;
  }
#pragma unroll
  for (int kk = 0; kk < FDIM; kk += 32) {
    v16h af = load_a_frag(sHid + arow * FDIM, kk, hi);
#pragma unroll
    for (int nt = 0; nt < 4; ++nt) {
      int n = c0 + nt * 16 + l15;
      v16h bf = load_b_frag(sW2t + n * FDIM, kk, hi);
      fc[nt] = __builtin_amdgcn_wmma_f32_16x16x32_f16(false, af, false, bf,
                                                      (short)0, fc[nt], false, false);
    }
  }

  // ---- filtered = nbr_feat * filters ; partial logits per row ----
  float p[8];
#pragma unroll
  for (int r = 0; r < 8; ++r) p[r] = 0.0f;
#pragma unroll
  for (int nt = 0; nt < 4; ++nt) {
    int col = c0 + nt * 16 + l15;
    float w = sNFw[col];
#pragma unroll
    for (int r = 0; r < 8; ++r) {
      int row = m0 + r + 8 * hi;
      float f = fc[nt][r] * sNF[row * FDIM + col];
      fc[nt][r] = f;                    // keep filtered in regs
      p[r] += f * w;
    }
  }
  // reduce over the 16 lanes that share each row (cols)
#pragma unroll
  for (int mask = 1; mask < 16; mask <<= 1) {
#pragma unroll
    for (int r = 0; r < 8; ++r) p[r] += __shfl_xor(p[r], mask, 32);
  }
  if (l15 == 0) {                       // lane 0: rows m0..m0+7, lane 16: rows m0+8..15
#pragma unroll
    for (int r = 0; r < 8; ++r) atomicAdd(&sLog[m0 + 8 * hi + r], p[r]);
  }
  __syncthreads();

  // ---- softmax over 64 neighbors ----
  if (tid < NDIM) {
    float mx = -3.4e38f;
#pragma unroll 8
    for (int j = 0; j < NDIM; ++j) mx = fmaxf(mx, sLog[j]);
    float s = 0.0f;
#pragma unroll 8
    for (int j = 0; j < NDIM; ++j) s += __expf(sLog[j] - mx);
    float a = __expf(sLog[tid] - mx) / s;
    sAtt[tid] = a;
    att_out[(size_t)ba * NDIM + tid] = a;
  }
  __syncthreads();

  // ---- output[col] = sum_row att[row] * filtered[row][col] ----
#pragma unroll
  for (int nt = 0; nt < 4; ++nt) {
    float acc = 0.0f;
#pragma unroll
    for (int r = 0; r < 8; ++r) acc += sAtt[m0 + r + 8 * hi] * fc[nt][r];
    acc += __shfl_xor(acc, 16, 32);     // partner lane holds same col, other 8 rows
    if (hi == 0) atomicAdd(&sOut[c0 + nt * 16 + l15], acc);
  }
  __syncthreads();
  if (tid < FDIM) out[(size_t)ba * FDIM + tid] = sOut[tid];
}

extern "C" void kernel_launch(void* const* d_in, const int* in_sizes, int n_in,
                              void* d_out, int out_size, void* d_ws, size_t ws_size,
                              hipStream_t stream) {
  const float*     feats = (const float*)d_in[0];
  const float*     rbf   = (const float*)d_in[1];
  const long long* nbrs  = (const long long*)d_in[2];
  const float*     W1    = (const float*)d_in[3];
  const float*     b1    = (const float*)d_in[4];
  const float*     W2    = (const float*)d_in[5];
  const float*     b2    = (const float*)d_in[6];
  const float*     nfw   = (const float*)d_in[7];
  float* out = (float*)d_out;                                  // [B,A,F]
  float* att = out + (size_t)BDIM * ADIM * FDIM;               // [B,A,N]

  dim3 grid(BDIM * ADIM);
  dim3 block(256);
  cfconv_kernel<<<grid, block, SMEM_BYTES, stream>>>(feats, rbf, nbrs, W1, b1, W2, b2,
                                                     nfw, out, att);
}
